// TransformerEncoder_23184233464125
// MI455X (gfx1250) — matrix operand order repaired and checked by reading, not compiled, not run on hardware
//
#include <hip/hip_runtime.h>
#include <hip/hip_bf16.h>

// Transformer encoder (L=2, B=4, S=2048, D=512, H=8, HD=64, FF=2048) for
// gfx1250. bf16 WMMA (v_wmma_f32_16x16x32_bf16) with f32 accumulation.
// Weights pre-converted to bf16 once per launch so GEMM weight tiles are
// staged into LDS by the Tensor Data Mover (tensor_load_to_lds +
// s_wait_tensorcnt); activations staged with cvt. QKV emit bf16 directly
// (V transposed per head) so flash attention streams pure bf16.

#define L_   2
#define B_   4
#define S_   2048
#define D_   512
#define H_   8
#define HD_  64
#define FF_  2048
#define BS_  (B_ * S_)

typedef __bf16 bf16_t;
typedef __attribute__((ext_vector_type(16))) __bf16 v16bf;
typedef __attribute__((ext_vector_type(8)))  float  v8f;
typedef __attribute__((ext_vector_type(4)))  unsigned int u32x4;
typedef __attribute__((ext_vector_type(8)))  int  i32x8;
typedef __attribute__((ext_vector_type(4)))  int  i32x4;

__device__ __forceinline__ v8f wmma_bf16(v16bf a, v16bf b, v8f c) {
  // 8 args: (neg_a, A, neg_b, B, c_mod, C, reuse_a, reuse_b)
  return __builtin_amdgcn_wmma_f32_16x16x32_bf16(false, a, false, b,
                                                 (short)0, c, false, false);
}

// ---------------------------------------------------------------------------
// f32 -> bf16 copy (weight pre-conversion, once per launch)
// ---------------------------------------------------------------------------
__global__ void __launch_bounds__(256) cvt_bf16_kernel(
    const float* __restrict__ src, bf16_t* __restrict__ dst, int n) {
  int i = blockIdx.x * 256 + threadIdx.x;
  if (i < n) dst[i] = (bf16_t)src[i];
}

// ---------------------------------------------------------------------------
// LayerNorm over D=512, one block (256 threads = 8 waves) per row.
// ---------------------------------------------------------------------------
__global__ void __launch_bounds__(256) ln512_kernel(
    const float* __restrict__ x, const float* __restrict__ g,
    const float* __restrict__ bb, float* __restrict__ out) {
  __shared__ float red[8];
  const int row = blockIdx.x, tid = threadIdx.x;
  const float* xr = x + (size_t)row * D_;
  float v0 = xr[tid], v1 = xr[tid + 256];

  float s = v0 + v1;
#pragma unroll
  for (int o = 16; o; o >>= 1) s += __shfl_xor(s, o, 32);
  if ((tid & 31) == 0) red[tid >> 5] = s;
  __syncthreads();
  float mean = ((red[0] + red[1]) + (red[2] + red[3])) +
               ((red[4] + red[5]) + (red[6] + red[7]));
  mean *= (1.f / (float)D_);

  float d0 = v0 - mean, d1 = v1 - mean;
  float vs = d0 * d0 + d1 * d1;
#pragma unroll
  for (int o = 16; o; o >>= 1) vs += __shfl_xor(vs, o, 32);
  __syncthreads();  // all reads of red done before reuse
  if ((tid & 31) == 0) red[tid >> 5] = vs;
  __syncthreads();
  float var = ((red[0] + red[1]) + (red[2] + red[3])) +
              ((red[4] + red[5]) + (red[6] + red[7]));
  var *= (1.f / (float)D_);

  const float inv = rsqrtf(var + 1e-5f);
  float* orow = out + (size_t)row * D_;
  orow[tid]       = d0 * inv * g[tid] + bb[tid];
  orow[tid + 256] = d1 * inv * g[tid + 256] + bb[tid + 256];
}

// ---------------------------------------------------------------------------
// C[M,N] = epilogue(A[M,K] @ W[K,N] + bias); A f32, W bf16 (TDM-staged).
// MODE 1: f32 +bias +residual   MODE 2: f32 +bias, ReLU
// MODE 3: bf16 +bias (row-major)
// MODE 4: bf16 +bias, stored transposed per head: Cout[(b*H+h)*HD+e][s]
// Block: 256 threads (8 waves), tile 64(M) x 128(N); wave tile 32x32.
// ---------------------------------------------------------------------------
template <int MODE>
__global__ void __launch_bounds__(256) gemm_bf16_kernel(
    const float* __restrict__ A, const bf16_t* __restrict__ W,
    const float* __restrict__ bias, const float* __restrict__ resid,
    void* __restrict__ Cout, int K, int N) {
  __shared__ bf16_t lA[64 * 32];    // [m][k]
  __shared__ bf16_t lB[32 * 128];   // [k][n] -- filled by TDM
  const int n0 = blockIdx.x * 128;
  const int m0 = blockIdx.y * 64;
  const int tid = threadIdx.x;
  const int wave = tid >> 5, lane = tid & 31;
  const int hi = lane >> 4, lr = lane & 15;
  const int wm = (wave & 1) * 32;
  const int wn = (wave >> 1) * 32;

  // Tensor DMA descriptor group 1 (constant): bf16 elements, tile 128x32,
  // tensor [K rows][N cols], dim0 stride = N elements.  (ISA 8.4)
  const unsigned int td0 = (unsigned int)N, td1 = (unsigned int)K;
  i32x8 g1v;
  g1v[0] = (int)(1u << 16);                               // data_size=1 (2B)
  g1v[1] = (int)((td0 & 0xFFFFu) << 16);                  // tensor_dim0 lo
  g1v[2] = (int)((td0 >> 16) | ((td1 & 0xFFFFu) << 16));  // td0 hi | td1 lo
  g1v[3] = (int)((td1 >> 16) | (128u << 16));             // td1 hi | tile_dim0
  g1v[4] = (int)32u;                                      // tile_dim1=32
  g1v[5] = (int)td0;                                      // dim0_stride lo
  g1v[6] = 0;                                             // stride hi bits
  g1v[7] = 0;
  const i32x4 z4 = {0, 0, 0, 0};
  const i32x8 z8 = {0, 0, 0, 0, 0, 0, 0, 0};
  const unsigned int lb_off = (unsigned int)(uintptr_t)(void*)lB;

  v8f acc[2][2] = {};

  for (int k0 = 0; k0 < K; k0 += 32) {
    // --- TDM: async copy W tile (32 x 128 bf16) global -> LDS (wave 0) ---
    if (tid < 32) {
      const unsigned long long ga =
          (unsigned long long)(uintptr_t)&W[(size_t)k0 * N + n0];
      u32x4 g0v;
      g0v[0] = 1u;                                        // count=1 (valid)
      g0v[1] = lb_off;                                    // LDS byte address
      g0v[2] = (unsigned int)ga;                          // global addr lo
      g0v[3] = (unsigned int)((ga >> 32) & 0x1FFFFFFu) | (2u << 30);  // type=2
      __builtin_amdgcn_tensor_load_to_lds(g0v, g1v, z4, z4, z8, 0);
    }
    // --- stage A tile (64 x 32 f32 -> bf16) with all threads meanwhile ---
    for (int i = tid; i < 64 * 32; i += 256) {
      int r = i >> 5, c = i & 31;
      const float* ap = &A[(size_t)(m0 + r) * K + (k0 + c)];
      lA[i] = (bf16_t)(*ap);
      if (k0 + 32 < K) __builtin_prefetch(ap + 32, 0, 3);  // global_prefetch_b8
    }
    if (tid < 32) __builtin_amdgcn_s_wait_tensorcnt(0);
    __syncthreads();

    v16bf af[2], bfr[2];
#pragma unroll
    for (int t = 0; t < 2; ++t) {
      const int row = wm + t * 16 + lr;
#pragma unroll
      for (int v = 0; v < 8; ++v) {
        // A fragment, contiguous k order: kb = 2v + 16*hi
        const int kb = 2 * v + (hi ? 16 : 0);
        af[t][2 * v]     = lA[row * 32 + kb];
        af[t][2 * v + 1] = lA[row * 32 + kb + 1];
      }
    }
#pragma unroll
    for (int t = 0; t < 2; ++t) {
      const int col = wn + t * 16 + lr;
#pragma unroll
      for (int v = 0; v < 8; ++v) {
        // B fragment, same k order as A: k = 2v + 16*hi, n = lane&15
        const int kb = 2 * v + (hi ? 16 : 0);
        bfr[t][2 * v]     = lB[kb * 128 + col];
        bfr[t][2 * v + 1] = lB[(kb + 1) * 128 + col];
      }
    }
#pragma unroll
    for (int i = 0; i < 2; ++i)
#pragma unroll
      for (int j = 0; j < 2; ++j) acc[i][j] = wmma_bf16(af[i], bfr[j], acc[i][j]);
    __syncthreads();
  }

#pragma unroll
  for (int i = 0; i < 2; ++i)
#pragma unroll
    for (int j = 0; j < 2; ++j) {
      const int col = n0 + wn + j * 16 + lr;
      const float bv = bias[col];
#pragma unroll
      for (int v = 0; v < 8; ++v) {
        const int row = m0 + wm + i * 16 + v + hi * 8;  // C/D layout
        float val = acc[i][j][v] + bv;
        if (MODE == 2) val = fmaxf(val, 0.f);
        if (MODE == 1) val += resid[(size_t)row * N + col];
        if (MODE <= 2) {
          ((float*)Cout)[(size_t)row * N + col] = val;
        } else if (MODE == 3) {
          ((bf16_t*)Cout)[(size_t)row * N + col] = (bf16_t)val;
        } else {  // MODE 4: V stored [b*H+h][e][s]
          const int bidx = row / S_, sidx = row % S_;
          const int hh = col / HD_, e = col % HD_;
          ((bf16_t*)Cout)[((size_t)(bidx * H_ + hh) * HD_ + e) * S_ + sidx] =
              (bf16_t)val;
        }
      }
    }
}

// ---------------------------------------------------------------------------
// Flash attention: one wave per 16-row Q tile per (b,h). Scores computed
// transposed (S^T = K_tile @ Q^T) so softmax reductions are per-lane + one
// shfl_xor(16). Online softmax; O^T = V^T @ P^T accumulated in registers.
// Q,K: bf16 [BS][D]; Vt: bf16 [(b*H+h)][HD][S] (pre-transposed).
// ---------------------------------------------------------------------------
__global__ void __launch_bounds__(128) attn_kernel(
    const bf16_t* __restrict__ Q, const bf16_t* __restrict__ Kp,
    const bf16_t* __restrict__ Vt, float* __restrict__ O) {
  __shared__ bf16_t pstage[4][32 * 16];   // per-wave P^T staging (bf16)
  const int wave = threadIdx.x >> 5, lane = threadIdx.x & 31;
  const int hi = lane >> 4, lr = lane & 15;
  const int wid = blockIdx.x * 4 + wave;          // 4096 q-tiles total
  const int qt = wid & 127;                        // S/16 tiles per head
  const int bh = wid >> 7;                         // 0..31
  const int b = bh >> 3, h = bh & 7;
  const size_t rowbase = (size_t)b * S_;
  const int fbase = h * HD_;
  bf16_t* pb = pstage[wave];

  // Q^T B-fragments (two feature halves of HD=64)
  v16bf qb[2];
  {
    const int qrow = qt * 16 + lr;
    const bf16_t* qp = Q + (rowbase + qrow) * D_ + fbase;
#pragma unroll
    for (int half = 0; half < 2; ++half)
#pragma unroll
      for (int v = 0; v < 8; ++v) {
        const int kb = 2 * v + (hi ? 16 : 0);
        qb[half][2 * v]     = qp[half * 32 + kb];
        qb[half][2 * v + 1] = qp[half * 32 + kb + 1];
      }
  }

  // V^T row pointers (A-matrix m = feature row, 4 subtiles)
  const bf16_t* vrow[4];
#pragma unroll
  for (int ft = 0; ft < 4; ++ft)
    vrow[ft] = Vt + ((size_t)(b * H_ + h) * HD_ + ft * 16 + lr) * S_;

  float runmax = -3.0e38f, runsum = 0.f;
  v8f oacc[4] = {};  // O^T, 4 feature subtiles of 16

  for (int kv0 = 0; kv0 < S_; kv0 += 32) {
    // --- scores S^T (two 16-kpos subtiles) ---
    v8f st[2];
#pragma unroll
    for (int t = 0; t < 2; ++t) {
      const int krow = kv0 + t * 16 + lr;
      const bf16_t* kp = Kp + (rowbase + krow) * D_ + fbase;
      if (kv0 + 32 < S_) __builtin_prefetch(kp + (size_t)32 * D_, 0, 3);
      v16bf ka0, ka1;
#pragma unroll
      for (int v = 0; v < 8; ++v) {
        const int kb = 2 * v + (hi ? 16 : 0);
        ka0[2 * v]     = kp[kb];
        ka0[2 * v + 1] = kp[kb + 1];
        ka1[2 * v]     = kp[32 + kb];
        ka1[2 * v + 1] = kp[32 + kb + 1];
      }
      v8f c = {};
      c = wmma_bf16(ka0, qb[0], c);
      c = wmma_bf16(ka1, qb[1], c);
      st[t] = c;
    }

    // --- online softmax (per-lane column q = lane&15); 4-way trees ---
    float mx[4];
#pragma unroll
    for (int i = 0; i < 4; ++i) {
      st[0][i]     *= 0.125f;  st[0][i + 4] *= 0.125f;
      st[1][i]     *= 0.125f;  st[1][i + 4] *= 0.125f;
      mx[i] = fmaxf(fmaxf(st[0][i], st[0][i + 4]),
                    fmaxf(st[1][i], st[1][i + 4]));
    }
    float cmax = fmaxf(fmaxf(mx[0], mx[1]), fmaxf(mx[2], mx[3]));
    cmax = fmaxf(cmax, __shfl_xor(cmax, 16, 32));
    const float newmax = fmaxf(runmax, cmax);
    const float corr = __expf(runmax - newmax);
    runmax = newmax;
    runsum *= corr;
#pragma unroll
    for (int ft = 0; ft < 4; ++ft)
#pragma unroll
      for (int v = 0; v < 8; ++v) oacc[ft][v] *= corr;

    float ps[4] = {0.f, 0.f, 0.f, 0.f};
#pragma unroll
    for (int t = 0; t < 2; ++t)
#pragma unroll
      for (int v = 0; v < 8; ++v) {
        const float p = __expf(st[t][v] - newmax);
        ps[v & 3] += p;
        const int kpos = t * 16 + v + hi * 8;  // C/D layout row
        pb[kpos * 16 + lr] = (bf16_t)p;
      }
    float psum = (ps[0] + ps[1]) + (ps[2] + ps[3]);
    psum += __shfl_xor(psum, 16, 32);
    runsum += psum;

    // CDNA5 split wait: our LDS stores must land before the B-frag reads
    asm volatile("s_wait_dscnt 0x0" ::: "memory");

    // --- P^T B-fragment (k = kpos 0..31, n = qrow) ---
    v16bf pfrag;
#pragma unroll
    for (int v = 0; v < 8; ++v) {
      const int kb = 2 * v + (hi ? 16 : 0);
      pfrag[2 * v]     = pb[kb * 16 + lr];
      pfrag[2 * v + 1] = pb[(kb + 1) * 16 + lr];
    }

    // --- O^T += V^T @ P^T (4 feature subtiles, contiguous bf16 rows) ---
#pragma unroll
    for (int ft = 0; ft < 4; ++ft) {
      const bf16_t* vp = vrow[ft] + kv0;
      if (kv0 + 32 < S_) __builtin_prefetch(vp + 32, 0, 3);
      v16bf va;
#pragma unroll
      for (int v = 0; v < 8; ++v) {
        const int kb = 2 * v + (hi ? 16 : 0);
        va[2 * v]     = vp[kb];
        va[2 * v + 1] = vp[kb + 1];
      }
      oacc[ft] = wmma_bf16(va, pfrag, oacc[ft]);
    }
  }

  const float inv = 1.f / runsum;
  const int qrow = qt * 16 + lr;
#pragma unroll
  for (int ft = 0; ft < 4; ++ft)
#pragma unroll
    for (int v = 0; v < 8; ++v) {
      const int feat = fbase + ft * 16 + v + hi * 8;  // C/D layout
      O[(rowbase + qrow) * D_ + feat] = oacc[ft][v] * inv;
    }
}

// ---------------------------------------------------------------------------
// Host-side orchestration (graph-capture safe: only async launches/copies).
// ---------------------------------------------------------------------------
extern "C" void kernel_launch(void* const* d_in, const int* in_sizes, int n_in,
                              void* d_out, int out_size, void* d_ws,
                              size_t ws_size, hipStream_t stream) {
  const float* q     = (const float*)d_in[0];
  const float* ln1_g = (const float*)d_in[1];
  const float* ln1_b = (const float*)d_in[2];
  const float* wq    = (const float*)d_in[3];
  const float* bq    = (const float*)d_in[4];
  const float* wk    = (const float*)d_in[5];
  const float* bk    = (const float*)d_in[6];
  const float* wv    = (const float*)d_in[7];
  const float* bv    = (const float*)d_in[8];
  const float* wo    = (const float*)d_in[9];
  const float* bo    = (const float*)d_in[10];
  const float* ln2_g = (const float*)d_in[11];
  const float* ln2_b = (const float*)d_in[12];
  const float* w1    = (const float*)d_in[13];
  const float* b1    = (const float*)d_in[14];
  const float* w2    = (const float*)d_in[15];
  const float* b2    = (const float*)d_in[16];

  const size_t ND  = (size_t)BS_ * D_;
  const size_t WDD = (size_t)L_ * D_ * D_;    // 524288
  const size_t WDF = (size_t)L_ * D_ * FF_;   // 2097152
  float* ws  = (float*)d_ws;
  float* x   = ws;            // running activation           16MB
  float* kn  = x  + ND;       // LN(k_in) == LN(v_in)         16MB
  float* qn  = kn + ND;       // LN(x); reused as y after ln2 16MB
  float* ao  = qn + ND;       // attention output (pre-wo)    16MB
  float* ff  = ao + ND;       // [BS, FF] hidden              64MB
  bf16_t* Qb  = (bf16_t*)(ff + (size_t)BS_ * FF_);  //         8MB
  bf16_t* Kb  = Qb + ND;                            //         8MB
  bf16_t* Vt  = Kb + ND;      // [(b*H+h)][HD][S]              8MB
  bf16_t* wqb = Vt + ND;      // bf16 weights                 12MB total
  bf16_t* wkb = wqb + WDD;
  bf16_t* wvb = wkb + WDD;
  bf16_t* wob = wvb + WDD;
  bf16_t* w1b = wob + WDD;
  bf16_t* w2b = w1b + WDF;
  // total ws use: ~164MB (assumed <= ws_size)

  (void)hipMemcpyAsync(x, q, ND * sizeof(float), hipMemcpyDeviceToDevice,
                       stream);

  // Pre-convert weights to bf16 (TDM copies raw bytes; also halves traffic)
  cvt_bf16_kernel<<<(int)((WDD + 255) / 256), 256, 0, stream>>>(wq, wqb, (int)WDD);
  cvt_bf16_kernel<<<(int)((WDD + 255) / 256), 256, 0, stream>>>(wk, wkb, (int)WDD);
  cvt_bf16_kernel<<<(int)((WDD + 255) / 256), 256, 0, stream>>>(wv, wvb, (int)WDD);
  cvt_bf16_kernel<<<(int)((WDD + 255) / 256), 256, 0, stream>>>(wo, wob, (int)WDD);
  cvt_bf16_kernel<<<(int)((WDF + 255) / 256), 256, 0, stream>>>(w1, w1b, (int)WDF);
  cvt_bf16_kernel<<<(int)((WDF + 255) / 256), 256, 0, stream>>>(w2, w2b, (int)WDF);

  const dim3 gD(D_ / 128, BS_ / 64);    // (4, 128)
  const dim3 gF(FF_ / 128, BS_ / 64);   // (16, 128)

  for (int l = 0; l < L_; ++l) {
    const float* g1 = ln1_g + (size_t)l * D_;
    const float* c1 = ln1_b + (size_t)l * D_;
    ln512_kernel<<<BS_, 256, 0, stream>>>(x, g1, c1, qn);
    ln512_kernel<<<BS_, 256, 0, stream>>>(q, g1, c1, kn);

    gemm_bf16_kernel<3><<<gD, 256, 0, stream>>>(
        qn, wqb + (size_t)l * D_ * D_, bq + (size_t)l * D_, nullptr, Qb, D_, D_);
    gemm_bf16_kernel<3><<<gD, 256, 0, stream>>>(
        kn, wkb + (size_t)l * D_ * D_, bk + (size_t)l * D_, nullptr, Kb, D_, D_);
    gemm_bf16_kernel<4><<<gD, 256, 0, stream>>>(
        kn, wvb + (size_t)l * D_ * D_, bv + (size_t)l * D_, nullptr, Vt, D_, D_);

    attn_kernel<<<(B_ * H_ * (S_ / 16)) / 4, 128, 0, stream>>>(Qb, Kb, Vt, ao);

    gemm_bf16_kernel<1><<<gD, 256, 0, stream>>>(
        ao, wob + (size_t)l * D_ * D_, bo + (size_t)l * D_, x, x, D_, D_);

    ln512_kernel<<<BS_, 256, 0, stream>>>(x, ln2_g + (size_t)l * D_,
                                          ln2_b + (size_t)l * D_, qn);
    gemm_bf16_kernel<2><<<gF, 256, 0, stream>>>(
        qn, w1b + (size_t)l * D_ * FF_, b1 + (size_t)l * FF_, nullptr, ff,
        D_, FF_);

    float* outp = (l == L_ - 1) ? (float*)d_out : x;
    gemm_bf16_kernel<1><<<gD, 256, 0, stream>>>(
        ff, w2b + (size_t)l * FF_ * D_, b2 + (size_t)l * D_, x, outp, FF_, D_);
  }
}
